// PairStackBlock_18159121727822
// MI455X (gfx1250) — compile-verified, matrix operand order checked by reading, not run on hardware
//
#include <hip/hip_runtime.h>
#include <hip/hip_bf16.h>

// ---------------------------------------------------------------------------
// AlphaFold pair-stack block for MI455X (gfx1250), bf16 WMMA everywhere.
//   z: [1,320,320,128] fp32.  All matmuls -> v_wmma_f32_16x16x32_bf16.
//   GEMM waves compute 16x64 output strips: one A fragment feeds 4 WMMAs
//   (4x operand reuse vs single-tile waves -> ~32 FLOP/byte from cache).
// Workspace layout (bytes), total need = 212,025,344 (~202 MiB):
//   zcur fp32 52.4M | zn bf16 26.2M | R3..R6 bf16 4x26.2M (a_t/q, b_t/k,
//   xn/v_t, gate; also contiguous 104.9M region for transition hidden) |
//   R7 bf16 26.2M (attn out) | tbias fp32 1.6M | mask_t 0.4M | weight pool.
// d_out (52.4M fp32) is used as scratch (x buffer / z-transposed) and is
// fully overwritten by the final transition stage.
// ---------------------------------------------------------------------------

#define NRES 320
#define NN   (NRES * NRES)   // 102400
#define CZ   128

using bf16 = __bf16;
typedef __attribute__((ext_vector_type(16))) __bf16 v16bf;
typedef __attribute__((ext_vector_type(8)))  __bf16 v8bf;
typedef __attribute__((ext_vector_type(8)))  float  v8f;

__device__ __forceinline__ unsigned short f2bf_s(float f) {
  unsigned u = __float_as_uint(f);
  u += 0x7FFFu + ((u >> 16) & 1u);           // round-to-nearest-even
  return (unsigned short)(u >> 16);
}
__device__ __forceinline__ bf16 f2bf(float f) {
  unsigned short h = f2bf_s(f);
  return __builtin_bit_cast(bf16, h);
}
__device__ __forceinline__ float bf2f(bf16 h) {
  unsigned s = (unsigned)__builtin_bit_cast(unsigned short, h);
  return __uint_as_float(s << 16);
}
__device__ __forceinline__ float sigmoidf(float x) { return 1.f / (1.f + __expf(-x)); }

// Load a 16x32 bf16 fragment (A layout; identical for B^T stored [n][k]).
// Per ISA 7.12.2: lanes 0-15 row r hold K {0..7,16..23}; lanes 16-31 hold
// K {8..15,24..31}.  Two 16-byte loads per lane.
__device__ __forceinline__ v16bf load_frag(const bf16* p, int ld) {
  int lane = threadIdx.x & 31;
  const bf16* base = p + (size_t)(lane & 15) * ld + ((lane >> 4) << 3);
  v8bf lo = *(const v8bf*)(base);
  v8bf hi = *(const v8bf*)(base + 16);
  return __builtin_shufflevector(lo, hi, 0,1,2,3,4,5,6,7,8,9,10,11,12,13,14,15);
}

__device__ __forceinline__ v8f wmma_bf(v16bf a, v16bf b, v8f c) {
  return __builtin_amdgcn_wmma_f32_16x16x32_bf16(false, a, false, b, (short)0, c,
                                                 false, false);
}

struct alignas(8) B4 { bf16 a, b, c, d; };

// ---------------- LayerNorm over last dim (128), fp32 in -> bf16 out -------
__global__ __launch_bounds__(256)
void ln_kernel(const float* __restrict__ src, const float* __restrict__ w,
               const float* __restrict__ b, bf16* __restrict__ dst, int M) {
  int wid  = (blockIdx.x * 256 + threadIdx.x) >> 5;
  int lane = threadIdx.x & 31;
  const float4 v = *(const float4*)(src + (size_t)wid * CZ + lane * 4);
  float s = v.x + v.y + v.z + v.w;
  float q = v.x * v.x + v.y * v.y + v.z * v.z + v.w * v.w;
  for (int off = 16; off; off >>= 1) { s += __shfl_xor(s, off); q += __shfl_xor(q, off); }
  float mu  = s * (1.f / 128.f);
  float var = q * (1.f / 128.f) - mu * mu;
  float rs  = rsqrtf(var + 1e-5f);
  float4 wv = *(const float4*)(w + lane * 4);
  float4 bv = *(const float4*)(b + lane * 4);
  B4 o;
  o.a = f2bf((v.x - mu) * rs * wv.x + bv.x);
  o.b = f2bf((v.y - mu) * rs * wv.y + bv.y);
  o.c = f2bf((v.z - mu) * rs * wv.z + bv.z);
  o.d = f2bf((v.w - mu) * rs * wv.w + bv.w);
  *(B4*)(dst + (size_t)wid * CZ + lane * 4) = o;
}

// ---------------- Weight transpose+convert: W[K,N] fp32 -> Wt[N,K] bf16 ----
__global__ __launch_bounds__(256)
void wt_kernel(const float* __restrict__ w, bf16* __restrict__ wt, int K, int N) {
  int idx = blockIdx.x * 256 + threadIdx.x;
  if (idx >= K * N) return;
  int n = idx / K, k = idx - n * K;
  wt[idx] = f2bf(w[(size_t)k * N + n]);
}

__global__ __launch_bounds__(256)
void copy_kernel(const float* __restrict__ src, float* __restrict__ dst) {
  int idx = blockIdx.x * 256 + threadIdx.x;
  ((float4*)dst)[idx] = ((const float4*)src)[idx];
}

__global__ __launch_bounds__(128)
void transpose_kernel(const float* __restrict__ src, float* __restrict__ dst) {
  int ij = blockIdx.x;           // 0..NN-1
  int i = ij / NRES, j = ij - i * NRES;
  dst[((size_t)j * NRES + i) * CZ + threadIdx.x] =
      src[((size_t)i * NRES + j) * CZ + threadIdx.x];
}

__global__ __launch_bounds__(256)
void maskT_kernel(const float* __restrict__ src, float* __restrict__ dst) {
  int idx = blockIdx.x * 256 + threadIdx.x;
  if (idx >= NN) return;
  int i = idx / NRES, j = idx - i * NRES;
  dst[(size_t)j * NRES + i] = src[idx];
}

// ---- tri-mul projection: dst_t[c][i*320+k or k*320+i] =
//      mask * (zn@Wp + bp) * sigmoid(zn@Wg + bg)   (bf16, channel-major).
//      Wave computes a 16x32 strip for BOTH gemms: 4 WMMAs share one A frag.
__global__ __launch_bounds__(256)
void tm_proj_kernel(const bf16* __restrict__ zn, const bf16* __restrict__ wpt,
                    const float* __restrict__ bp, const bf16* __restrict__ wgt,
                    const float* __restrict__ bg, const float* __restrict__ mask,
                    bf16* __restrict__ dst, int M, int swap) {
  int wid = (blockIdx.x * 256 + threadIdx.x) >> 5;
  int mt = wid >> 2, st = wid & 3;              // 4 strips of 32 cols (N=128)
  int m0 = mt * 16, n0 = st * 32;
  v8f aP0 = (v8f)0.0f, aP1 = (v8f)0.0f, aG0 = (v8f)0.0f, aG1 = (v8f)0.0f;
#pragma unroll
  for (int k0 = 0; k0 < 128; k0 += 32) {
    v16bf af = load_frag(zn + (size_t)m0 * CZ + k0, CZ);
    aP0 = wmma_bf(af, load_frag(wpt + (size_t)n0 * 128 + k0, 128), aP0);
    aP1 = wmma_bf(af, load_frag(wpt + (size_t)(n0 + 16) * 128 + k0, 128), aP1);
    aG0 = wmma_bf(af, load_frag(wgt + (size_t)n0 * 128 + k0, 128), aG0);
    aG1 = wmma_bf(af, load_frag(wgt + (size_t)(n0 + 16) * 128 + k0, 128), aG1);
  }
  int lane = threadIdx.x & 31, hf = lane >> 4, n = lane & 15;
  int c0 = n0 + n, c1 = n0 + 16 + n;
  float bP0 = bp[c0], bP1 = bp[c1], bG0 = bg[c0], bG1 = bg[c1];
#pragma unroll
  for (int v = 0; v < 8; v++) {
    int m = m0 + v + 8 * hf;
    float mk = mask[m];
    int r1 = m / NRES, r2 = m - r1 * NRES;
    int mi = swap ? (r2 * NRES + r1) : m;
    dst[(size_t)c0 * M + mi] = f2bf(mk * (aP0[v] + bP0) * sigmoidf(aG0[v] + bG0));
    dst[(size_t)c1 * M + mi] = f2bf(mk * (aP1[v] + bP1) * sigmoidf(aG1[v] + bG1));
  }
}

// ---- per-channel triangle GEMM: X_c = A_c[I,K] @ B_c[J,K]^T -> x[i,j,c].
//      Wave computes a 16x64 j-strip: A frag reused by 4 WMMAs per k-step. --
__global__ __launch_bounds__(256)
void tri_gemm_kernel(const bf16* __restrict__ at, const bf16* __restrict__ bt,
                     float* __restrict__ x) {
  int wid = (blockIdx.x * 256 + threadIdx.x) >> 5;
  int js = wid % 5, it = (wid / 5) % 20, c = wid / 100;   // 5 j-strips of 64
  const bf16* A = at + (size_t)c * NN;
  const bf16* B = bt + (size_t)c * NN;
  int i0 = it * 16, j0 = js * 64;
  v8f acc[4];
#pragma unroll
  for (int t = 0; t < 4; t++) acc[t] = (v8f)0.0f;
#pragma unroll
  for (int k0 = 0; k0 < NRES; k0 += 32) {
    v16bf af = load_frag(A + (size_t)i0 * NRES + k0, NRES);
#pragma unroll
    for (int t = 0; t < 4; t++)
      acc[t] = wmma_bf(af, load_frag(B + (size_t)(j0 + 16 * t) * NRES + k0, NRES),
                       acc[t]);
  }
  int lane = threadIdx.x & 31, hf = lane >> 4, n = lane & 15;
#pragma unroll
  for (int t = 0; t < 4; t++) {
#pragma unroll
    for (int v = 0; v < 8; v++) {
      int i = i0 + v + 8 * hf, j = j0 + 16 * t + n;
      x[((size_t)i * NRES + j) * CZ + c] = acc[t][v];
    }
  }
}

// ---- tri-mul output: zcur += (xn@zw + zb) * sigmoid(zn@gw + gb).
//      16x32 strip for both gemms; 4 WMMAs per k-step. ----------------------
__global__ __launch_bounds__(256)
void tm_out_kernel(const bf16* __restrict__ xn, const bf16* __restrict__ zn,
                   const bf16* __restrict__ zwt, const float* __restrict__ zb,
                   const bf16* __restrict__ gwt, const float* __restrict__ gb,
                   float* __restrict__ zcur) {
  int wid = (blockIdx.x * 256 + threadIdx.x) >> 5;
  int mt = wid >> 2, st = wid & 3;
  int m0 = mt * 16, n0 = st * 32;
  v8f aX0 = (v8f)0.0f, aX1 = (v8f)0.0f, aG0 = (v8f)0.0f, aG1 = (v8f)0.0f;
#pragma unroll
  for (int k0 = 0; k0 < 128; k0 += 32) {
    v16bf xf = load_frag(xn + (size_t)m0 * CZ + k0, CZ);
    v16bf zf = load_frag(zn + (size_t)m0 * CZ + k0, CZ);
    aX0 = wmma_bf(xf, load_frag(zwt + (size_t)n0 * 128 + k0, 128), aX0);
    aX1 = wmma_bf(xf, load_frag(zwt + (size_t)(n0 + 16) * 128 + k0, 128), aX1);
    aG0 = wmma_bf(zf, load_frag(gwt + (size_t)n0 * 128 + k0, 128), aG0);
    aG1 = wmma_bf(zf, load_frag(gwt + (size_t)(n0 + 16) * 128 + k0, 128), aG1);
  }
  int lane = threadIdx.x & 31, hf = lane >> 4, n = lane & 15;
  int c0 = n0 + n, c1 = n0 + 16 + n;
  float bZ0 = zb[c0], bZ1 = zb[c1], bG0 = gb[c0], bG1 = gb[c1];
#pragma unroll
  for (int v = 0; v < 8; v++) {
    int m = m0 + v + 8 * hf;
    zcur[(size_t)m * CZ + c0] += (aX0[v] + bZ0) * sigmoidf(aG0[v] + bG0);
    zcur[(size_t)m * CZ + c1] += (aX1[v] + bZ1) * sigmoidf(aG1[v] + bG1);
  }
}

// ---- generic GEMM -> bf16: 16x64 strip/wave; act 0=none 1=sigmoid 2=relu;
//      vt=1 stores [i][c][y] (per-row-block transposed, for attention V) ----
__global__ __launch_bounds__(256)
void gemm_bf_kernel(const bf16* __restrict__ A, const bf16* __restrict__ wt,
                    const float* __restrict__ bias, bf16* __restrict__ dst,
                    int K, int N, float scale, int act, int vt) {
  int wid = (blockIdx.x * 256 + threadIdx.x) >> 5;
  int nstrips = N >> 6;
  int mt = wid / nstrips, st = wid - mt * nstrips;
  int m0 = mt * 16, n0 = st * 64;
  v8f acc[4];
#pragma unroll
  for (int t = 0; t < 4; t++) acc[t] = (v8f)0.0f;
  for (int k0 = 0; k0 < K; k0 += 32) {
    const bf16* ap = A + (size_t)m0 * K + k0;
    if (k0 + 64 < K) __builtin_prefetch(ap + 64, 0, 0);   // global_prefetch_b8
    v16bf af = load_frag(ap, K);
#pragma unroll
    for (int t = 0; t < 4; t++)
      acc[t] = wmma_bf(af, load_frag(wt + (size_t)(n0 + 16 * t) * K + k0, K),
                       acc[t]);
  }
  int lane = threadIdx.x & 31, hf = lane >> 4, n = lane & 15;
#pragma unroll
  for (int t = 0; t < 4; t++) {
    int c = n0 + 16 * t + n;
    float bb = bias ? bias[c] : 0.f;
#pragma unroll
    for (int v = 0; v < 8; v++) {
      int m = m0 + v + 8 * hf;
      float val = acc[t][v] + bb;
      if (act == 1) val = sigmoidf(val);
      else if (act == 2) val = fmaxf(val, 0.f);
      val *= scale;
      size_t idx;
      if (vt) {
        int i = m / NRES, y = m - i * NRES;
        idx = (size_t)i * (CZ * NRES) + (size_t)c * NRES + y;
      } else {
        idx = (size_t)m * N + c;
      }
      dst[idx] = f2bf(val);
    }
  }
}

// ---- generic GEMM + bias (+ mask) + residual add, fp32 out (N==128) -------
__global__ __launch_bounds__(256)
void gemm_add_kernel(const bf16* __restrict__ A, const bf16* __restrict__ wt,
                     const float* __restrict__ bias, const float* __restrict__ mask,
                     const float* __restrict__ src, float* __restrict__ dst, int K) {
  int wid = (blockIdx.x * 256 + threadIdx.x) >> 5;
  int mt = wid >> 1, st = wid & 1;               // 2 strips of 64 (N=128)
  int m0 = mt * 16, n0 = st * 64;
  v8f acc[4];
#pragma unroll
  for (int t = 0; t < 4; t++) acc[t] = (v8f)0.0f;
  for (int k0 = 0; k0 < K; k0 += 32) {
    const bf16* ap = A + (size_t)m0 * K + k0;
    if (k0 + 64 < K) __builtin_prefetch(ap + 64, 0, 0);
    v16bf af = load_frag(ap, K);
#pragma unroll
    for (int t = 0; t < 4; t++)
      acc[t] = wmma_bf(af, load_frag(wt + (size_t)(n0 + 16 * t) * K + k0, K),
                       acc[t]);
  }
  int lane = threadIdx.x & 31, hf = lane >> 4, n = lane & 15;
#pragma unroll
  for (int t = 0; t < 4; t++) {
    int c = n0 + 16 * t + n;
    float bb = bias[c];
#pragma unroll
    for (int v = 0; v < 8; v++) {
      int m = m0 + v + 8 * hf;
      float val = acc[t][v] + bb;
      if (mask) val *= mask[m];
      dst[(size_t)m * CZ + c] = src[(size_t)m * CZ + c] + val;
    }
  }
}

// ---- triangle bias: tbias[m][h] = zn[m] . tb_w[:,h]  ----------------------
__global__ __launch_bounds__(256)
void tb_kernel(const bf16* __restrict__ zn, const float* __restrict__ w,
               float* __restrict__ tb) {
  int idx = blockIdx.x * 256 + threadIdx.x;   // m*4 + h
  int m = idx >> 2, h = idx & 3;
  float s = 0.f;
  for (int c = 0; c < 128; c++) s += bf2f(zn[(size_t)m * CZ + c]) * w[c * 4 + h];
  tb[idx] = s;
}

// ---- flash attention per (row i, head h, 16-query tile), K step 32 --------
__global__ __launch_bounds__(256)
void attn_kernel(const bf16* __restrict__ qb, const bf16* __restrict__ kb,
                 const bf16* __restrict__ vt, const bf16* __restrict__ gb,
                 const float* __restrict__ tbias, const float* __restrict__ mask,
                 bf16* __restrict__ ob) {
  __shared__ bf16 ptile[8 * 16 * 32];
  int warp = threadIdx.x >> 5;
  int wid  = blockIdx.x * 8 + warp;
  int xt = wid % 20, h = (wid / 20) & 3, i = wid / 80;
  int x0 = xt * 16;
  int lane = threadIdx.x & 31, hf = lane >> 4, n = lane & 15;
  bf16* myP = ptile + warp * 512;

  v16bf qf = load_frag(qb + ((size_t)i * NRES + x0) * CZ + h * 32, CZ);
  v8f o0 = (v8f)0.0f, o1 = (v8f)0.0f;
  float rmax[8], rsum[8];
#pragma unroll
  for (int v = 0; v < 8; v++) { rmax[v] = -1e30f; rsum[v] = 0.f; }

  for (int y0 = 0; y0 < NRES; y0 += 32) {
    v16bf kf0 = load_frag(kb + ((size_t)i * NRES + y0) * CZ + h * 32, CZ);
    v16bf kf1 = load_frag(kb + ((size_t)i * NRES + y0 + 16) * CZ + h * 32, CZ);
    v8f zero = (v8f)0.0f;
    v8f s0 = wmma_bf(qf, kf0, zero);
    v8f s1 = wmma_bf(qf, kf1, zero);
    float mb0 = 1e9f * (mask[(size_t)i * NRES + y0 + n] - 1.f);
    float mb1 = 1e9f * (mask[(size_t)i * NRES + y0 + 16 + n] - 1.f);
    float p0[8], p1[8];
#pragma unroll
    for (int v = 0; v < 8; v++) {
      int x = x0 + v + 8 * hf;
      s0[v] += mb0 + tbias[((size_t)x * NRES + y0 + n) * 4 + h];
      s1[v] += mb1 + tbias[((size_t)x * NRES + y0 + 16 + n) * 4 + h];
      float tm = fmaxf(s0[v], s1[v]);
      for (int off = 8; off; off >>= 1) tm = fmaxf(tm, __shfl_xor(tm, off));
      float mn = fmaxf(rmax[v], tm);
      float sc = __expf(rmax[v] - mn);
      o0[v] *= sc; o1[v] *= sc; rsum[v] *= sc; rmax[v] = mn;
      p0[v] = __expf(s0[v] - mn);
      p1[v] = __expf(s1[v] - mn);
      float rs = p0[v] + p1[v];
      for (int off = 8; off; off >>= 1) rs += __shfl_xor(rs, off);
      rsum[v] += rs;
    }
#pragma unroll
    for (int v = 0; v < 8; v++) {
      int r = v + 8 * hf;
      myP[r * 32 + n]      = f2bf(p0[v]);
      myP[r * 32 + 16 + n] = f2bf(p1[v]);
    }
    asm volatile("" ::: "memory");   // LDS is in-order per wave (DScnt)
    v16bf pf  = load_frag(myP, 32);
    v16bf vf0 = load_frag(vt + (size_t)i * (CZ * NRES) + (size_t)(h * 32) * NRES + y0, NRES);
    v16bf vf1 = load_frag(vt + (size_t)i * (CZ * NRES) + (size_t)(h * 32 + 16) * NRES + y0, NRES);
    o0 = wmma_bf(pf, vf0, o0);
    o1 = wmma_bf(pf, vf1, o1);
  }
#pragma unroll
  for (int v = 0; v < 8; v++) {
    int x = x0 + v + 8 * hf;
    float inv = 1.f / rsum[v];
    size_t base = ((size_t)i * NRES + x) * CZ + h * 32;
    ob[base + n]      = f2bf(o0[v] * inv * bf2f(gb[base + n]));
    ob[base + 16 + n] = f2bf(o1[v] * inv * bf2f(gb[base + 16 + n]));
  }
}

// ---------------------------------------------------------------------------
extern "C" void kernel_launch(void* const* d_in, const int* in_sizes, int n_in,
                              void* d_out, int out_size, void* d_ws, size_t ws_size,
                              hipStream_t stream) {
  (void)in_sizes; (void)n_in; (void)out_size;
  constexpr size_t OFF_ZN = 52428800, OFF_R3 = 78643200, OFF_R4 = 104857600,
                   OFF_R5 = 131072000, OFF_R6 = 157286400, OFF_R7 = 183500800,
                   OFF_TB = 209715200, OFF_MT = 211353600, OFF_WP = 211763200,
                   WS_NEED = 212025344;
  if (ws_size < WS_NEED) return;

  char* w = (char*)d_ws;
  float* zcur  = (float*)w;
  bf16*  zn    = (bf16*)(w + OFF_ZN);
  bf16*  r3    = (bf16*)(w + OFF_R3);   // a_t / q / h[0:]
  bf16*  r4    = (bf16*)(w + OFF_R4);   // b_t / k
  bf16*  r5    = (bf16*)(w + OFF_R5);   // xn / v_t
  bf16*  r6    = (bf16*)(w + OFF_R6);   // gate
  bf16*  r7    = (bf16*)(w + OFF_R7);   // attn out
  float* tbias = (float*)(w + OFF_TB);
  float* maskt = (float*)(w + OFF_MT);
  bf16*  wpool = (bf16*)(w + OFF_WP);   // 8 slots x 16384 bf16
  float* xbuf  = (float*)d_out;         // scratch until final stage
  float* zt    = (float*)d_out;

  const float* Zin  = (const float*)d_in[0];
  const float* MASK = (const float*)d_in[1];
  auto in = [&](int i) { return (const float*)d_in[i]; };
  auto slot = [&](int i) { return wpool + (size_t)i * 16384; };
  auto wt = [&](const float* W, bf16* Wt, int K, int N) {
    wt_kernel<<<(K * N + 255) / 256, 256, 0, stream>>>(W, Wt, K, N);
  };

  copy_kernel<<<NN * CZ / 1024, 256, 0, stream>>>(Zin, zcur);

  // ---- triangle multiplicative update (outgoing: swap=0, incoming: swap=1)
  auto tri_mul = [&](int b, int swap) {
    wt(in(b + 2),  slot(0), 128, 128);  // a_p
    wt(in(b + 4),  slot(1), 128, 128);  // a_g
    wt(in(b + 6),  slot(2), 128, 128);  // b_p
    wt(in(b + 8),  slot(3), 128, 128);  // b_g
    wt(in(b + 12), slot(4), 128, 128);  // z_w
    wt(in(b + 14), slot(5), 128, 128);  // g_w
    ln_kernel<<<NN / 8, 256, 0, stream>>>(zcur, in(b), in(b + 1), zn, NN);
    tm_proj_kernel<<<3200, 256, 0, stream>>>(zn, slot(0), in(b + 3), slot(1),
                                             in(b + 5), MASK, r3, NN, swap);
    tm_proj_kernel<<<3200, 256, 0, stream>>>(zn, slot(2), in(b + 7), slot(3),
                                             in(b + 9), MASK, r4, NN, swap);
    tri_gemm_kernel<<<1600, 256, 0, stream>>>(r3, r4, xbuf);
    ln_kernel<<<NN / 8, 256, 0, stream>>>(xbuf, in(b + 10), in(b + 11), r5, NN);
    tm_out_kernel<<<3200, 256, 0, stream>>>(r5, zn, slot(4), in(b + 13), slot(5),
                                            in(b + 15), zcur);
  };

  // ---- triangle attention on zb with mask mk -------------------------------
  auto tri_att = [&](int b, float* zb, const float* mk) {
    wt(in(b + 2), slot(0), 128, 128);   // q_w
    wt(in(b + 3), slot(1), 128, 128);   // k_w
    wt(in(b + 4), slot(2), 128, 128);   // v_w
    wt(in(b + 6), slot(3), 128, 128);   // g_w
    wt(in(b + 8), slot(4), 128, 128);   // o_w
    ln_kernel<<<NN / 8, 256, 0, stream>>>(zb, in(b), in(b + 1), zn, NN);
    gemm_bf_kernel<<<1600, 256, 0, stream>>>(zn, slot(0), nullptr, r3, 128, 128,
                                             0.17677669529663687f, 0, 0);  // q
    gemm_bf_kernel<<<1600, 256, 0, stream>>>(zn, slot(1), nullptr, r4, 128, 128,
                                             1.f, 0, 0);                    // k
    gemm_bf_kernel<<<1600, 256, 0, stream>>>(zn, slot(2), nullptr, r5, 128, 128,
                                             1.f, 0, 1);                    // v_t
    gemm_bf_kernel<<<1600, 256, 0, stream>>>(zn, slot(3), in(b + 7), r6, 128, 128,
                                             1.f, 1, 0);                    // gate
    tb_kernel<<<NN * 4 / 256, 256, 0, stream>>>(zn, in(b + 5), tbias);
    attn_kernel<<<3200, 256, 0, stream>>>(r3, r4, r5, r6, tbias, mk, r7);
    gemm_add_kernel<<<1600, 256, 0, stream>>>(r7, slot(4), in(b + 9), nullptr,
                                              zb, zb, 128);
  };

  tri_mul(2, 0);                         // tmo (outgoing)
  tri_mul(18, 1);                        // tmi (incoming)
  tri_att(34, zcur, MASK);               // tas (starting node)

  transpose_kernel<<<NN, 128, 0, stream>>>(zcur, zt);
  maskT_kernel<<<(NN + 255) / 256, 256, 0, stream>>>(MASK, maskt);
  tri_att(44, zt, maskt);                // tae (ending node, on transposed z)
  transpose_kernel<<<NN, 128, 0, stream>>>(zt, zcur);

  // ---- transition: d_out = zcur + mask * (relu(ln(zcur)@w1+b1)@w2 + b2) ---
  wt(in(56), slot(0), 128, 512);         // w1t: [512,128] (slots 0-3)
  wt(in(58), slot(4), 512, 128);         // w2t: [128,512] (slots 4-7)
  ln_kernel<<<NN / 8, 256, 0, stream>>>(zcur, in(54), in(55), zn, NN);
  gemm_bf_kernel<<<6400, 256, 0, stream>>>(zn, slot(0), in(57), r3, 128, 512,
                                           1.f, 2, 0);   // hidden, relu, 104.9MB
  gemm_add_kernel<<<1600, 256, 0, stream>>>(r3, slot(4), in(59), MASK, zcur,
                                            (float*)d_out, 512);
}
